// Classification_3058016714941
// MI455X (gfx1250) — matrix operand order repaired and checked by reading, not compiled, not run on hardware
//
#include <hip/hip_runtime.h>
#include <hip/hip_bf16.h>

#define SEQ_LEN    512
#define BATCH      4096
#define HID        16
#define NUM_LABELS 256

typedef __attribute__((ext_vector_type(16))) _Float16 v16h;
typedef __attribute__((ext_vector_type(8)))  float    v8f;

// Hardware reciprocal (v_rcp_f32): avoids the IEEE div_scale/div_fixup chain.
__device__ __forceinline__ float fast_rcp(float x) {
  return __builtin_amdgcn_rcpf(x);
}
__device__ __forceinline__ float sigmoid_f(float x) {
  return fast_rcp(1.0f + __expf(-x));
}
__device__ __forceinline__ float tanh_f(float x) {
  float ax = fabsf(x);
  float e  = __expf(-2.0f * ax);
  float r  = (1.0f - e) * fast_rcp(1.0f + e);
  return copysignf(r, x);
}

// A-operand tile (16x32 f16, K=16 real + 16 zero padding):
// lane l holds row m = l%16; halves 0..7 = K = (l/16)*8 .. +7; halves 8..15 = 0.
__device__ __forceinline__ v16h pack_a8(float4 a, float4 b) {
  v16h r = {};
  r[0] = (_Float16)a.x; r[1] = (_Float16)a.y; r[2] = (_Float16)a.z; r[3] = (_Float16)a.w;
  r[4] = (_Float16)b.x; r[5] = (_Float16)b.y; r[6] = (_Float16)b.z; r[7] = (_Float16)b.w;
  return r;
}

// B-operand column (lanes 0..15 hold K=0..15; lanes 16..31 are K=16..31 -> zero pad)
__device__ __forceinline__ v16h pack_b16(float4 a, float4 b, float4 c, float4 d) {
  v16h r;
  r[0]  = (_Float16)a.x; r[1]  = (_Float16)a.y; r[2]  = (_Float16)a.z; r[3]  = (_Float16)a.w;
  r[4]  = (_Float16)b.x; r[5]  = (_Float16)b.y; r[6]  = (_Float16)b.z; r[7]  = (_Float16)b.w;
  r[8]  = (_Float16)c.x; r[9]  = (_Float16)c.y; r[10] = (_Float16)c.z; r[11] = (_Float16)c.w;
  r[12] = (_Float16)d.x; r[13] = (_Float16)d.y; r[14] = (_Float16)d.z; r[15] = (_Float16)d.w;
  return r;
}

// One wave per block; each wave owns a 16-wide batch tile and runs the full
// 512-step recurrence with weights resident in VGPRs as WMMA A-operands.
// Gate formulation: D(gate,batch) = W(gate,hid) x h^T(hid,batch).
__global__ void __launch_bounds__(32)
lstm_cls_kernel(const int*   __restrict__ x,
                const float* __restrict__ emb,
                const float* __restrict__ w_ih,
                const float* __restrict__ w_hh,
                const float* __restrict__ b_ih,
                const float* __restrict__ b_hh,
                const float* __restrict__ w_fc,
                const float* __restrict__ b_fc,
                float*       __restrict__ out)
{
  const int lane = threadIdx.x;      // 0..31 (wave32)
  const int b0   = blockIdx.x * 16;  // batch tile base
  const int half = lane >> 4;        // 0 or 1
  const int lrow = lane & 15;

  // ---- preload weights (A operands) and fused biases (C layout) ----
  v16h a_ih[4], a_hh[4];
  v8f  bias[4];
#pragma unroll
  for (int t = 0; t < 4; ++t) {
    const float4* wi = (const float4*)(w_ih + (size_t)(t * 16 + lrow) * HID + half * 8);
    const float4* wh = (const float4*)(w_hh + (size_t)(t * 16 + lrow) * HID + half * 8);
    a_ih[t] = pack_a8(wi[0], wi[1]);
    a_hh[t] = pack_a8(wh[0], wh[1]);
#pragma unroll
    for (int j = 0; j < 8; ++j) {
      const int g = t * 16 + j + half * 8;  // C layout: vgpr j -> row m = j + 8*half
      bias[t][j] = b_ih[g] + b_hh[g];
    }
  }

  // LSTM state in C/D layout: lane holds (hid m = j + 8*half, batch n = lrow)
  float cst[8], hf[8];
#pragma unroll
  for (int j = 0; j < 8; ++j) { cst[j] = 0.0f; hf[j] = 0.0f; }
  v16h hB = {};  // h^T as B operand (starts at zero state)
  const v16h zv = {};

  for (int s = 0; s < SEQ_LEN; ++s) {
    // warm the index stream a few steps ahead (global_prefetch_b8)
    if (s + 4 < SEQ_LEN)
      __builtin_prefetch(x + (size_t)(s + 4) * BATCH + b0 + lrow, 0, 0);

    // ---- x_t^T as B operand: lane holds emb[x[s][b0+lrow]][half*8 .. +7] in elems 0..7 ----
    v16h bx = {};
    {
      const int idx = x[(size_t)s * BATCH + b0 + lrow];
      const float4* r = (const float4*)(emb + (size_t)idx * HID + half * 8);
      float4 e0 = r[0], e1 = r[1];
      if (idx == 0) {  // padding_idx=0 -> frozen zero row
        e0 = e1 = make_float4(0.f, 0.f, 0.f, 0.f);
      }
      bx = pack_a8(e0, e1);
    }

    // ---- gates = W_ih x_t^T + (b_ih+b_hh) + W_hh h^T : 8 WMMAs ----
    v8f acc[4];
#pragma unroll
    for (int t = 0; t < 4; ++t) {
      acc[t] = __builtin_amdgcn_wmma_f32_16x16x32_f16(
          false, a_ih[t], false, bx, (short)0, bias[t], false, false);
      acc[t] = __builtin_amdgcn_wmma_f32_16x16x32_f16(
          false, a_hh[t], false, hB, (short)0, acc[t], false, false);
    }

    // ---- activations + state update (torch gate order i,f,g,o) ----
#pragma unroll
    for (int j = 0; j < 8; ++j) {
      const float ig = sigmoid_f(acc[0][j]);
      const float fg = sigmoid_f(acc[1][j]);
      const float gg = tanh_f(acc[2][j]);
      const float og = sigmoid_f(acc[3][j]);
      cst[j] = fg * cst[j] + ig * gg;
      hf[j]  = og * tanh_f(cst[j]);
    }

    // ---- rebuild h^T B operand: in-lane converts + 8 xor-16 shuffles ----
    v16h hn;
#pragma unroll
    for (int j = 0; j < 8; ++j) {
      const float oth = __shfl_xor(hf[j], 16, 32);  // partner half's hid dims
      hn[j]     = (_Float16)hf[j];
      hn[j + 8] = (_Float16)0.0f;
    }

    hB = hn;
  }

  // ---- classifier: logits(label,batch) = w_fc x h^T + b_fc, 16 tiles ----
#pragma unroll 1
  for (int t = 0; t < NUM_LABELS / 16; ++t) {
    const float4* wf = (const float4*)(w_fc + (size_t)(t * 16 + lrow) * HID + half * 8);
    const v16h a_fc = pack_a8(wf[0], wf[1]);
    v8f cc;
#pragma unroll
    for (int j = 0; j < 8; ++j) cc[j] = b_fc[t * 16 + j + half * 8];
    cc = __builtin_amdgcn_wmma_f32_16x16x32_f16(
        false, a_fc, false, hB, (short)0, cc, false, false);
    // store: out[b0 + lrow][t*16 + half*8 + j], contiguous 8 floats per lane
    float* op = out + (size_t)(b0 + lrow) * NUM_LABELS + t * 16 + half * 8;
    ((float4*)op)[0] = make_float4(cc[0], cc[1], cc[2], cc[3]);
    ((float4*)op)[1] = make_float4(cc[4], cc[5], cc[6], cc[7]);
  }
}

// In-place log_softmax: one wave per batch row (256 logits -> 8 per lane).
__global__ void __launch_bounds__(32)
log_softmax_kernel(float* __restrict__ out)
{
  const int lane = threadIdx.x;
  float* p = out + (size_t)blockIdx.x * NUM_LABELS + lane * 8;
  float4 a = ((const float4*)p)[0];
  float4 b = ((const float4*)p)[1];
  float v[8] = {a.x, a.y, a.z, a.w, b.x, b.y, b.z, b.w};

  float m = v[0];
#pragma unroll
  for (int j = 1; j < 8; ++j) m = fmaxf(m, v[j]);
#pragma unroll
  for (int off = 16; off >= 1; off >>= 1) m = fmaxf(m, __shfl_xor(m, off, 32));

  float s = 0.0f;
#pragma unroll
  for (int j = 0; j < 8; ++j) s += __expf(v[j] - m);
#pragma unroll
  for (int off = 16; off >= 1; off >>= 1) s += __shfl_xor(s, off, 32);

  const float lse = m + __logf(s);
  ((float4*)p)[0] = make_float4(v[0] - lse, v[1] - lse, v[2] - lse, v[3] - lse);
  ((float4*)p)[1] = make_float4(v[4] - lse, v[5] - lse, v[6] - lse, v[7] - lse);
}

extern "C" void kernel_launch(void* const* d_in, const int* in_sizes, int n_in,
                              void* d_out, int out_size, void* d_ws, size_t ws_size,
                              hipStream_t stream) {
  (void)in_sizes; (void)n_in; (void)d_ws; (void)ws_size; (void)out_size;
  const int*   x    = (const int*)  d_in[0];
  const float* emb  = (const float*)d_in[1];
  const float* w_ih = (const float*)d_in[2];
  const float* w_hh = (const float*)d_in[3];
  const float* b_ih = (const float*)d_in[4];
  const float* b_hh = (const float*)d_in[5];
  const float* w_fc = (const float*)d_in[6];
  const float* b_fc = (const float*)d_in[7];
  float* out = (float*)d_out;

  // 256 independent waves (one per 16-batch tile) -> fills the WGP array
  lstm_cls_kernel<<<BATCH / 16, 32, 0, stream>>>(
      x, emb, w_ih, w_hh, b_ih, b_hh, w_fc, b_fc, out);
  log_softmax_kernel<<<BATCH, 32, 0, stream>>>(out);
}
